// LSTM_85864986182202
// MI455X (gfx1250) — matrix operand order repaired and checked by reading, not compiled, not run on hardware
//
#include <hip/hip_runtime.h>

typedef __attribute__((ext_vector_type(16))) __bf16 v16bf;
typedef __attribute__((ext_vector_type(8)))  float  v8f;

#define BH 65536        // B*H = 64*1024 (elements)
#define NWG 64          // H/16 column tiles

__device__ __forceinline__ unsigned short f2bf(float f) {
    unsigned int u = __float_as_uint(f);
    u = (u + 0x7FFFu + ((u >> 16) & 1u)) >> 16;   // round-to-nearest-even
    return (unsigned short)u;
}
__device__ __forceinline__ float fsigmoid(float x) { return 1.0f / (1.0f + __expf(-x)); }
__device__ __forceinline__ float ftanh(float x)    { return 1.0f - 2.0f / (__expf(2.0f * x) + 1.0f); }

// B-fragment: 32 contiguous bytes per lane (fragment-packed weights)
__device__ __forceinline__ v16bf load_b(const unsigned short* p) {
    union { uint4 u[2]; v16bf v; } t;
    t.u[0] = *(const uint4*)(p);
    t.u[1] = *(const uint4*)(p + 8);
    return t.v;
}
// A-fragment from a row-major bf16 row: halfs {base..+7} and {base+16..+23}
__device__ __forceinline__ v16bf load_a(const unsigned short* p) {
    union { uint4 u[2]; v16bf v; } t;
    t.u[0] = *(const uint4*)(p);
    t.u[1] = *(const uint4*)(p + 16);
    return t.v;
}

__device__ __forceinline__ void grid_barrier(unsigned* bar, unsigned target) {
    __syncthreads();
    if (threadIdx.x == 0) {
        __threadfence();
        __hip_atomic_fetch_add(bar, 1u, __ATOMIC_RELEASE, __HIP_MEMORY_SCOPE_AGENT);
        while (__hip_atomic_load(bar, __ATOMIC_ACQUIRE, __HIP_MEMORY_SCOPE_AGENT) < target) {
            __builtin_amdgcn_s_sleep(2);
        }
    }
    __syncthreads();
}

// ---- weight packing into WMMA B-fragment order ------------------------------
// packed[((l*4+g)*64 + ntile)*32 + ktile] is a 512-half fragment:
//   lane (0..31) holds 16 halfs: n = lane&15, K = ktile*32 + (j&7) + (j&8 ? 16 : 0) + (lane&16 ? 8 : 0)
__global__ void pack_rnn(const float* __restrict__ Wih, const float* __restrict__ Whh,
                         unsigned short* __restrict__ pWih, unsigned short* __restrict__ pWhh) {
    unsigned tid = blockIdx.x * 256u + threadIdx.x;     // 2^25 threads total
    const float* W = (tid & (1u << 24)) ? Whh : Wih;
    unsigned short* out = (tid & (1u << 24)) ? pWhh : pWih;
    unsigned i    = tid & 0xFFFFFFu;
    unsigned j    = i & 15u;
    unsigned lane = (i >> 4) & 31u;
    unsigned kt   = (i >> 9) & 31u;
    unsigned nt   = (i >> 14) & 63u;
    unsigned g    = (i >> 20) & 3u;
    unsigned l    = (i >> 22) & 3u;
    unsigned n    = lane & 15u;
    unsigned K    = kt * 32u + (j & 7u) + ((j & 8u) ? 16u : 0u) + ((lane & 16u) ? 8u : 0u);
    unsigned row  = g * 1024u + nt * 16u + n;
    out[i] = f2bf(W[((size_t)l * 4096u + row) * 1024u + K]);
}

__global__ void pack_fc(const float* __restrict__ Wfc, unsigned short* __restrict__ pWfc) {
    unsigned i = blockIdx.x * 256u + threadIdx.x;       // 2^19 threads
    unsigned j    = i & 15u;
    unsigned lane = (i >> 4) & 31u;
    unsigned kt   = (i >> 9) & 31u;
    unsigned nt   = (i >> 14) & 31u;
    unsigned n    = lane & 15u;
    unsigned K    = kt * 32u + (j & 7u) + ((j & 8u) ? 16u : 0u) + ((lane & 16u) ? 8u : 0u);
    unsigned row  = nt * 16u + n;
    pWfc[i] = f2bf(Wfc[(size_t)row * 1024u + K]);
}

// x: [B,T,D] fp32  ->  xbf: [T,B,D] bf16 (row-major rows so A-frags are 2x16B loads)
__global__ void pack_x(const float* __restrict__ x, unsigned short* __restrict__ xbf) {
    unsigned i = blockIdx.x * 256u + threadIdx.x;       // 2^24 threads
    unsigned k = i & 1023u;
    unsigned t = (i >> 10) & 255u;
    unsigned b = i >> 18;
    xbf[((size_t)t * 64u + b) * 1024u + k] = f2bf(x[((size_t)b * 256u + t) * 1024u + k]);
}

__global__ void init_state(unsigned short* hbf, float* cbuf, unsigned* bar) {
    unsigned i = blockIdx.x * 256u + threadIdx.x;
    if (i < 262144u) { hbf[i] = 0; cbuf[i] = 0.0f; }   // L*B*H
    if (i == 0) *bar = 0u;
}

// ---- persistent LSTM kernel: 64 WGs, each owns one 16-wide h/c column tile ---
__global__ __launch_bounds__(256) void lstm_persist(
    const unsigned short* __restrict__ xbf,
    const float* __restrict__ b_ih, const float* __restrict__ b_hh,
    const unsigned short* __restrict__ pWih, const unsigned short* __restrict__ pWhh,
    unsigned short* __restrict__ hbf, float* __restrict__ cbuf, unsigned* bar)
{
    __shared__ float sGate[4][64][16];   // 16 KB gate exchange (f32)
    __shared__ float sBias[16][16];      // [l*4+g][n] hoisted bias sums

    const int tid   = threadIdx.x;
    const int lane  = tid & 31;
    const int w     = tid >> 5;          // wave 0..7 (wave32)
    const int mtile = w & 3;             // batch row tile 0..3
    const int gp    = (w >> 2) << 1;     // gate pair base: 0 or 2
    const int hi    = (lane >> 4) & 1;
    const int col   = blockIdx.x;        // h column tile 0..63
    unsigned rounds = 0;

    // hoist bias sums: constant over time
    {
        int l = tid >> 6, g = (tid >> 4) & 3, n = tid & 15;
        int o = l * 4096 + g * 1024 + col * 16 + n;
        sBias[tid >> 4][n] = b_ih[o] + b_hh[o];
    }
    __syncthreads();

    // per-lane A-fragment row offset (halfs), row stride 1024 halfs
    const int arow = (mtile * 16 + (lane & 15)) * 1024 + hi * 8;

    for (int t = 0; t < 256; ++t) {
        for (int l = 0; l < 4; ++l) {
            v8f acc0 = {}; v8f acc1 = {};
            for (int half = 0; half < 2; ++half) {
                const unsigned short* src;
                const unsigned short* pw;
                if (half == 0) {
                    src = (l == 0) ? xbf + (size_t)t * BH
                                   : hbf + (size_t)(l - 1) * BH;
                    pw = pWih;
                } else {
                    src = hbf + (size_t)l * BH;
                    pw = pWhh;
                }
                const unsigned short* ap = src + arow;
                const unsigned short* bp =
                    pw + ((((size_t)l * 4 + gp) * 64 + col) * 32) * 512 + lane * 16;
                #pragma unroll 4
                for (int kt = 0; kt < 32; ++kt) {
                    v16bf A  = load_a(ap + kt * 32);
                    v16bf B0 = load_b(bp + (size_t)kt * 512);
                    v16bf B1 = load_b(bp + (size_t)kt * 512 + 2048u * 512u); // next gate
                    acc0 = __builtin_amdgcn_wmma_f32_16x16x32_bf16(
                               false, A, false, B0, (short)0, acc0, false, false);
                    acc1 = __builtin_amdgcn_wmma_f32_16x16x32_bf16(
                               false, A, false, B1, (short)0, acc1, false, false);
                }
            }
            // C/D layout: VGPR r -> M = r + (lane>=16 ? 8 : 0), N = lane&15
            #pragma unroll
            for (int r = 0; r < 8; ++r) {
                int m = mtile * 16 + r + hi * 8;
                int n = lane & 15;
                sGate[gp    ][m][n] = acc0[r];
                sGate[gp + 1][m][n] = acc1[r];
            }
            __syncthreads();
            // fused LSTM cell pointwise (gate order i,f,g,o)
            for (int e = tid; e < 1024; e += 256) {
                int m = e >> 4, n = e & 15;
                int gcol = col * 16 + n;
                float zi = sGate[0][m][n] + sBias[l * 4 + 0][n];
                float zf = sGate[1][m][n] + sBias[l * 4 + 1][n];
                float zg = sGate[2][m][n] + sBias[l * 4 + 2][n];
                float zo = sGate[3][m][n] + sBias[l * 4 + 3][n];
                float ig = fsigmoid(zi), fg = fsigmoid(zf), og = fsigmoid(zo), gg = ftanh(zg);
                size_t idx = (size_t)l * BH + (size_t)m * 1024 + gcol;
                float cn = fg * cbuf[idx] + ig * gg;
                cbuf[idx] = cn;
                hbf[idx]  = f2bf(og * ftanh(cn));
            }
            ++rounds;
            grid_barrier(bar, rounds * (unsigned)NWG);
        }
    }
}

// ---- final FC: out[64,512] = h[L-1] @ W_fc^T + b_fc --------------------------
__global__ __launch_bounds__(128) void fc_kernel(
    const unsigned short* __restrict__ hbf, const unsigned short* __restrict__ pWfc,
    const float* __restrict__ b_fc, float* __restrict__ out)
{
    const int tid  = threadIdx.x;
    const int lane = tid & 31;
    const int w    = tid >> 5;        // mtile 0..3
    const int hi   = (lane >> 4) & 1;
    const int col  = blockIdx.x;      // 0..31 output column tile
    v8f acc = {};
    const unsigned short* ap =
        hbf + (size_t)3 * BH + (w * 16 + (lane & 15)) * 1024 + hi * 8;
    const unsigned short* bp = pWfc + ((size_t)col * 32) * 512 + lane * 16;
    #pragma unroll 4
    for (int kt = 0; kt < 32; ++kt) {
        v16bf A = load_a(ap + kt * 32);
        v16bf B = load_b(bp + (size_t)kt * 512);
        acc = __builtin_amdgcn_wmma_f32_16x16x32_bf16(
                  false, A, false, B, (short)0, acc, false, false);
    }
    #pragma unroll
    for (int r = 0; r < 8; ++r) {
        int b = w * 16 + r + hi * 8;
        int n = col * 16 + (lane & 15);
        out[(size_t)b * 512 + n] = acc[r] + b_fc[n];
    }
}

extern "C" void kernel_launch(void* const* d_in, const int* in_sizes, int n_in,
                              void* d_out, int out_size, void* d_ws, size_t ws_size,
                              hipStream_t stream) {
    (void)in_sizes; (void)n_in; (void)out_size; (void)ws_size;
    const float* x   = (const float*)d_in[0];
    const float* Wih = (const float*)d_in[1];
    const float* Whh = (const float*)d_in[2];
    const float* bih = (const float*)d_in[3];
    const float* bhh = (const float*)d_in[4];
    const float* Wfc = (const float*)d_in[5];
    const float* bfc = (const float*)d_in[6];

    char* ws = (char*)d_ws;
    unsigned short* pWih = (unsigned short*)(ws);                    // 33,554,432 B
    unsigned short* pWhh = (unsigned short*)(ws + 33554432);         // 33,554,432 B
    unsigned short* pWfc = (unsigned short*)(ws + 67108864);         //  1,048,576 B
    unsigned short* xbf  = (unsigned short*)(ws + 68157440);         // 33,554,432 B
    unsigned short* hbf  = (unsigned short*)(ws + 101711872);        //    524,288 B
    float*          cbuf = (float*)(ws + 102236160);                 //  1,048,576 B
    unsigned*       bar  = (unsigned*)(ws + 103284736);

    init_state<<<1024, 256, 0, stream>>>(hbf, cbuf, bar);
    pack_rnn<<<131072, 256, 0, stream>>>(Wih, Whh, pWih, pWhh);
    pack_fc<<<2048, 256, 0, stream>>>(Wfc, pWfc);
    pack_x<<<65536, 256, 0, stream>>>(x, xbf);
    lstm_persist<<<NWG, 256, 0, stream>>>(xbf, bih, bhh, pWih, pWhh, hbf, cbuf, bar);
    fc_kernel<<<32, 128, 0, stream>>>(hbf, pWfc, bfc, (float*)d_out);
}
